// EncoderSimpleConv_80015240725036
// MI455X (gfx1250) — matrix operand order repaired and checked by reading, not compiled, not run on hardware
//
#include <hip/hip_runtime.h>

#define NN 32768
#define DG 32
#define DH 64

typedef __attribute__((ext_vector_type(16))) _Float16 v16h;
typedef __attribute__((ext_vector_type(8)))  _Float16 h8;
typedef __attribute__((ext_vector_type(4)))  _Float16 h4;
typedef __attribute__((ext_vector_type(8)))  float    v8f;
typedef __attribute__((ext_vector_type(4)))  float    f4;

__device__ __forceinline__ v16h frag_cat(h8 a, h8 b) {
  v16h r;
#pragma unroll
  for (int i = 0; i < 8; ++i) { r[i] = a[i]; r[i + 8] = b[i]; }
  return r;
}

// A fragment (16x32 f16 tile) from row-major f16, leading dim `ld` halves.
// lanes 0-15: row m0+lane, K {0..7, 16..23}; lanes 16-31: same rows, K {8..15, 24..31}
__device__ __forceinline__ v16h load_A(const _Float16* p, int ld, int m0, int k0, int lane) {
  int m  = m0 + (lane & 15);
  int hi = lane >> 4;
  h8 a = *(const h8*)(p + m * ld + k0 + 8 * hi);
  h8 b = *(const h8*)(p + m * ld + k0 + 16 + 8 * hi);
  return frag_cat(a, b);
}

// B fragment (32x16 f16 tile) from "Bt" row-major [N][K] (row n = column n of B).
// lanes 0-15 hold K {0..7, 16..23} of column n0+lane; lanes 16-31 hold K {8..15, 24..31}
__device__ __forceinline__ v16h load_B(const _Float16* bt, int ld, int n0, int k0, int lane) {
  int n  = n0 + (lane & 15);
  int hi = lane >> 4;
  h8 a = *(const h8*)(bt + n * ld + k0 + 8 * hi);
  h8 b = *(const h8*)(bt + n * ld + k0 + 16 + 8 * hi);
  return frag_cat(a, b);
}

template<int MT, int NT, int KT>
__device__ __forceinline__ void wave_gemm(const _Float16* A, int lda,
                                          const _Float16* Bt, int ldb,
                                          int lane, v8f acc[MT][NT]) {
#pragma unroll
  for (int mt = 0; mt < MT; ++mt)
#pragma unroll
    for (int nt = 0; nt < NT; ++nt) {
#pragma unroll
      for (int i = 0; i < 8; ++i) acc[mt][nt][i] = 0.0f;
    }
#pragma unroll
  for (int kt = 0; kt < KT; ++kt) {
#pragma unroll
    for (int mt = 0; mt < MT; ++mt) {
      v16h a = load_A(A, lda, mt * 16, kt * 32, lane);
#pragma unroll
      for (int nt = 0; nt < NT; ++nt) {
        v16h b = load_B(Bt, ldb, nt * 16, kt * 32, lane);
        acc[mt][nt] = __builtin_amdgcn_wmma_f32_16x16x32_f16(
            false, a, false, b, (short)0, acc[mt][nt], false, false);
      }
    }
  }
}

__device__ __forceinline__ float nan2num(float v) {
  if (v != v) return 0.0f;
  if (v >  3.402823466e38f) return  3.402823466e38f;
  if (v < -3.402823466e38f) return -3.402823466e38f;
  return v;
}

// ---------------- prep: transpose weights to f16 [dout][64], q_seed ----------------
__device__ void transpose_w(const float* W, int dout, _Float16* dst, int tid) {
  for (int idx = tid; idx < 64 * dout; idx += 256) {
    int n = idx / 64, k = idx % 64;        // dst[n*64+k] = W[k*dout+n]
    dst[idx] = (_Float16)W[k * dout + n];
  }
}

__global__ void prep_kernel(const float* __restrict__ w_lin,
                            const float* __restrict__ enc_in_w,
                            const float* __restrict__ enc_out_w,
                            const float* __restrict__ enc_ff_w,
                            const float* __restrict__ pma_lin_w,
                            const float* __restrict__ pma_in_w,
                            const float* __restrict__ pma_in_b,
                            const float* __restrict__ pma_seed,
                            _Float16* __restrict__ wt,
                            float* __restrict__ qseed) {
  int tid = threadIdx.x;
  transpose_w(w_lin,     64,  wt + 0,     tid);
  transpose_w(enc_in_w,  192, wt + 4096,  tid);
  transpose_w(enc_out_w, 64,  wt + 16384, tid);
  transpose_w(enc_ff_w,  64,  wt + 20480, tid);
  transpose_w(pma_lin_w, 64,  wt + 24576, tid);
  transpose_w(pma_in_w,  192, wt + 28672, tid);
  if (tid < 64) {
    float a = pma_in_b[tid];
    for (int k = 0; k < 64; ++k) a += pma_seed[k] * pma_in_w[k * 192 + tid];
    qseed[tid] = a;
  }
}

// ---------------- lin1: hfeat = x @ w_lin + b_lin  (WMMA) ----------------
__global__ __launch_bounds__(256) void lin1_kernel(const float* __restrict__ x,
                                                   const float* __restrict__ b_lin,
                                                   const _Float16* __restrict__ wt,
                                                   float* __restrict__ hfeat) {
  __shared__ alignas(16) _Float16 a16[8][DG * DH];
  int lane = threadIdx.x & 31, wid = threadIdx.x >> 5;
  int r0 = (blockIdx.x * 8 + wid) * DG;
  const float* xr = x + (size_t)(r0 + lane) * DH;
#pragma unroll
  for (int c = 0; c < DH; c += 4) {
    f4 v = *(const f4*)(xr + c);
    h4 hv; hv[0] = (_Float16)v[0]; hv[1] = (_Float16)v[1];
    hv[2] = (_Float16)v[2]; hv[3] = (_Float16)v[3];
    *(h4*)&a16[wid][lane * DH + c] = hv;
  }
  __syncthreads();
  v8f acc[2][4];
  wave_gemm<2, 4, 2>(a16[wid], DH, wt, DH, lane, acc);
#pragma unroll
  for (int mt = 0; mt < 2; ++mt)
#pragma unroll
    for (int nt = 0; nt < 4; ++nt) {
      int n = nt * 16 + (lane & 15);
#pragma unroll
      for (int v = 0; v < 8; ++v) {
        int m = mt * 16 + v + 8 * (lane >> 4);
        hfeat[(size_t)(r0 + m) * DH + n] = acc[mt][nt][v] + b_lin[n];
      }
    }
}

// ---------------- main: per-node set transformer, one wave32 per node ----------------
struct SM {
  float    xf[DG * DH];    // gathered set (f32), later residual H, later Kp
  float    vf[DG * DH];    // Vp (f32)
  float    sc[DG * DG];    // scores / scalar scratch (1024 f32)
  _Float16 a16[DG * DH];   // f16 activation staging
  _Float16 k16[DG * DH];   // f16 staging (Kmat, O', Henc)
  _Float16 q16[DG * DH];   // f16 staging (Q, H)
  _Float16 vt16[DH * DG];  // V transposed (column-major rows)
};

__global__ __launch_bounds__(32) void set_enc_kernel(
    const float* __restrict__ hfeat, const _Float16* __restrict__ wt,
    const float* __restrict__ qseed, const int* __restrict__ edge_src,
    const float* __restrict__ enc_in_b, const float* __restrict__ enc_out_b,
    const float* __restrict__ enc_ff_b, const float* __restrict__ pma_lin_b,
    const float* __restrict__ pma_in_b, const float* __restrict__ pma_seed,
    const float* __restrict__ pma_out_w, const float* __restrict__ pma_out_b,
    const float* __restrict__ pma_ff_w, const float* __restrict__ pma_ff_b,
    const float* __restrict__ dec_in_w, const float* __restrict__ dec_in_b,
    const float* __restrict__ dec_out_w, const float* __restrict__ dec_out_b,
    const float* __restrict__ dec_ff_w, const float* __restrict__ dec_ff_b,
    float* __restrict__ out) {
  __shared__ alignas(16) SM s;
  const int lane = threadIdx.x;
  const int node = blockIdx.x;
  const _Float16* wt_enc_in  = wt + 4096;
  const _Float16* wt_enc_out = wt + 16384;
  const _Float16* wt_enc_ff  = wt + 20480;
  const _Float16* wt_pma_lin = wt + 24576;
  const _Float16* wt_pma_in  = wt + 28672;

  // gather neighbor rows: lane = slot j
  {
    int src = edge_src[node * DG + lane];
    const float* hr = hfeat + (size_t)src * DH;
#pragma unroll
    for (int c = 0; c < DH; c += 4) {
      f4 v = *(const f4*)(hr + c);
      *(f4*)&s.xf[lane * DH + c] = v;
      h4 hv; hv[0] = (_Float16)v[0]; hv[1] = (_Float16)v[1];
      hv[2] = (_Float16)v[2]; hv[3] = (_Float16)v[3];
      *(h4*)&s.a16[lane * DH + c] = hv;
    }
  }
  __syncthreads();

  v8f acc[2][4];
  const int nl = lane & 15, mh = 8 * (lane >> 4);

  // ---- encoder SAB: Q, K, V ----
  wave_gemm<2, 4, 2>(s.a16, DH, wt_enc_in, DH, lane, acc);
#pragma unroll
  for (int mt = 0; mt < 2; ++mt)
#pragma unroll
    for (int nt = 0; nt < 4; ++nt) { int n = nt * 16 + nl;
#pragma unroll
      for (int v = 0; v < 8; ++v) { int m = mt * 16 + v + mh;
        s.q16[m * DH + n] = (_Float16)(acc[mt][nt][v] + enc_in_b[n]); } }
  wave_gemm<2, 4, 2>(s.a16, DH, wt_enc_in + 64 * DH, DH, lane, acc);
#pragma unroll
  for (int mt = 0; mt < 2; ++mt)
#pragma unroll
    for (int nt = 0; nt < 4; ++nt) { int n = nt * 16 + nl;
#pragma unroll
      for (int v = 0; v < 8; ++v) { int m = mt * 16 + v + mh;
        s.k16[m * DH + n] = (_Float16)(acc[mt][nt][v] + enc_in_b[64 + n]); } }
  wave_gemm<2, 4, 2>(s.a16, DH, wt_enc_in + 128 * DH, DH, lane, acc);
#pragma unroll
  for (int mt = 0; mt < 2; ++mt)
#pragma unroll
    for (int nt = 0; nt < 4; ++nt) { int n = nt * 16 + nl;
#pragma unroll
      for (int v = 0; v < 8; ++v) { int m = mt * 16 + v + mh;
        s.vt16[n * DG + m] = (_Float16)(acc[mt][nt][v] + enc_in_b[128 + n]); } }
  __syncthreads();

  // ---- S = Q K^T / 8 ----
  {
    v8f a2[2][2];
    wave_gemm<2, 2, 2>(s.q16, DH, s.k16, DH, lane, a2);
#pragma unroll
    for (int mt = 0; mt < 2; ++mt)
#pragma unroll
      for (int nt = 0; nt < 2; ++nt) { int n = nt * 16 + nl;
#pragma unroll
        for (int v = 0; v < 8; ++v) { int m = mt * 16 + v + mh;
          s.sc[m * DG + n] = a2[mt][nt][v] * 0.125f; } }
  }
  __syncthreads();

  // ---- row softmax (lane = row), write A f16 [32][32] ----
  {
    float e[DG];
    const float* row = s.sc + lane * DG;
    float mx = -3.402823466e38f;
#pragma unroll
    for (int j = 0; j < DG; ++j) mx = fmaxf(mx, row[j]);
    float sum = 0.0f;
#pragma unroll
    for (int j = 0; j < DG; ++j) { e[j] = __expf(row[j] - mx); sum += e[j]; }
    float inv = 1.0f / sum;
#pragma unroll
    for (int j = 0; j < DG; ++j) s.a16[lane * DG + j] = (_Float16)(e[j] * inv);
  }
  __syncthreads();

  // ---- O' = A @ V ----
  wave_gemm<2, 4, 1>(s.a16, DG, s.vt16, DG, lane, acc);
#pragma unroll
  for (int mt = 0; mt < 2; ++mt)
#pragma unroll
    for (int nt = 0; nt < 4; ++nt) { int n = nt * 16 + nl;
#pragma unroll
      for (int v = 0; v < 8; ++v) { int m = mt * 16 + v + mh;
        s.k16[m * DH + n] = (_Float16)acc[mt][nt][v]; } }
  __syncthreads();

  // ---- O = O' Wo + bo ; H = O + X ----
  wave_gemm<2, 4, 2>(s.k16, DH, wt_enc_out, DH, lane, acc);
#pragma unroll
  for (int mt = 0; mt < 2; ++mt)
#pragma unroll
    for (int nt = 0; nt < 4; ++nt) { int n = nt * 16 + nl;
#pragma unroll
      for (int v = 0; v < 8; ++v) { int m = mt * 16 + v + mh;
        float h = acc[mt][nt][v] + enc_out_b[n] + s.xf[m * DH + n];
        s.xf[m * DH + n] = h; s.q16[m * DH + n] = (_Float16)h; } }
  __syncthreads();

  // ---- Henc = H + relu(H Wff + bff) ----
  wave_gemm<2, 4, 2>(s.q16, DH, wt_enc_ff, DH, lane, acc);
#pragma unroll
  for (int mt = 0; mt < 2; ++mt)
#pragma unroll
    for (int nt = 0; nt < 4; ++nt) { int n = nt * 16 + nl;
#pragma unroll
      for (int v = 0; v < 8; ++v) { int m = mt * 16 + v + mh;
        float t = fmaxf(acc[mt][nt][v] + enc_ff_b[n], 0.0f);
        s.k16[m * DH + n] = (_Float16)(s.xf[m * DH + n] + t); } }
  __syncthreads();

  // ---- PMA: XP = relu(Henc Wl + bl) ----
  wave_gemm<2, 4, 2>(s.k16, DH, wt_pma_lin, DH, lane, acc);
#pragma unroll
  for (int mt = 0; mt < 2; ++mt)
#pragma unroll
    for (int nt = 0; nt < 4; ++nt) { int n = nt * 16 + nl;
#pragma unroll
      for (int v = 0; v < 8; ++v) { int m = mt * 16 + v + mh;
        s.a16[m * DH + n] = (_Float16)fmaxf(acc[mt][nt][v] + pma_lin_b[n], 0.0f); } }
  __syncthreads();

  // ---- Kp, Vp (f32 to LDS) ----
  wave_gemm<2, 4, 2>(s.a16, DH, wt_pma_in + 64 * DH, DH, lane, acc);
#pragma unroll
  for (int mt = 0; mt < 2; ++mt)
#pragma unroll
    for (int nt = 0; nt < 4; ++nt) { int n = nt * 16 + nl;
#pragma unroll
      for (int v = 0; v < 8; ++v) { int m = mt * 16 + v + mh;
        s.xf[m * DH + n] = acc[mt][nt][v] + pma_in_b[64 + n]; } }
  wave_gemm<2, 4, 2>(s.a16, DH, wt_pma_in + 128 * DH, DH, lane, acc);
#pragma unroll
  for (int mt = 0; mt < 2; ++mt)
#pragma unroll
    for (int nt = 0; nt < 4; ++nt) { int n = nt * 16 + nl;
#pragma unroll
      for (int v = 0; v < 8; ++v) { int m = mt * 16 + v + mh;
        s.vf[m * DH + n] = acc[mt][nt][v] + pma_in_b[128 + n]; } }
  __syncthreads();

  // ---- PMA attention (1 seed over 32 keys), lane = key j ----
  float* sa = s.sc;        // 32
  float* ov = s.sc + 32;   // 64
  float* pv = s.sc + 96;   // 64
  float* vv = s.sc + 160;  // 64
  float* hv = s.sc + 224;  // 64
  {
    float sacc = 0.0f;
    for (int d = 0; d < DH; ++d) sacc += qseed[d] * s.xf[lane * DH + d];
    sacc *= 0.125f;
    float mx = sacc;
#pragma unroll
    for (int off = 16; off > 0; off >>= 1) mx = fmaxf(mx, __shfl_xor(mx, off, 32));
    float e = __expf(sacc - mx);
    float sm = e;
#pragma unroll
    for (int off = 16; off > 0; off >>= 1) sm += __shfl_xor(sm, off, 32);
    sa[lane] = e / sm;
  }
  __syncthreads();
  {
    float o0 = 0.0f, o1 = 0.0f;
    for (int j = 0; j < DG; ++j) {
      float aj = sa[j];
      o0 += aj * s.vf[j * DH + lane];
      o1 += aj * s.vf[j * DH + lane + 32];
    }
    ov[lane] = o0; ov[lane + 32] = o1;
  }
  __syncthreads();
  float p0 = pma_out_b[lane]      + pma_seed[lane];
  float p1 = pma_out_b[lane + 32] + pma_seed[lane + 32];
  for (int d = 0; d < DH; ++d) {
    float od = ov[d];
    p0 += od * pma_out_w[d * DH + lane];
    p1 += od * pma_out_w[d * DH + lane + 32];
  }
  pv[lane] = p0; pv[lane + 32] = p1;
  __syncthreads();
  {
    float t0 = pma_ff_b[lane], t1 = pma_ff_b[lane + 32];
    for (int d = 0; d < DH; ++d) {
      float pd = pv[d];
      t0 += pd * pma_ff_w[d * DH + lane];
      t1 += pd * pma_ff_w[d * DH + lane + 32];
    }
    p0 += fmaxf(t0, 0.0f); p1 += fmaxf(t1, 0.0f);
  }
  __syncthreads();
  pv[lane] = p0; pv[lane + 32] = p1;   // p2
  __syncthreads();

  // ---- decoder SAB (1x1): softmax over single key == 1 -> attn out = V ----
  {
    float v0 = dec_in_b[128 + lane], v1 = dec_in_b[128 + lane + 32];
    for (int d = 0; d < DH; ++d) {
      float pd = pv[d];
      v0 += pd * dec_in_w[d * 192 + 128 + lane];
      v1 += pd * dec_in_w[d * 192 + 128 + lane + 32];
    }
    vv[lane] = v0; vv[lane + 32] = v1;
  }
  __syncthreads();
  float hh0 = dec_out_b[lane]      + p0;
  float hh1 = dec_out_b[lane + 32] + p1;
  for (int d = 0; d < DH; ++d) {
    float vd = vv[d];
    hh0 += vd * dec_out_w[d * DH + lane];
    hh1 += vd * dec_out_w[d * DH + lane + 32];
  }
  hv[lane] = hh0; hv[lane + 32] = hh1;
  __syncthreads();
  {
    float t0 = dec_ff_b[lane], t1 = dec_ff_b[lane + 32];
    for (int d = 0; d < DH; ++d) {
      float hd = hv[d];
      t0 += hd * dec_ff_w[d * DH + lane];
      t1 += hd * dec_ff_w[d * DH + lane + 32];
    }
    hh0 += fmaxf(t0, 0.0f); hh1 += fmaxf(t1, 0.0f);
  }
  out[(size_t)node * DH + lane]      = fmaxf(nan2num(hh0), 0.0f);
  out[(size_t)node * DH + lane + 32] = fmaxf(nan2num(hh1), 0.0f);
}

extern "C" void kernel_launch(void* const* d_in, const int* in_sizes, int n_in,
                              void* d_out, int out_size, void* d_ws, size_t ws_size,
                              hipStream_t stream) {
  const float* x         = (const float*)d_in[0];
  const float* w_lin     = (const float*)d_in[1];
  const float* b_lin     = (const float*)d_in[2];
  const float* enc_in_w  = (const float*)d_in[3];
  const float* enc_in_b  = (const float*)d_in[4];
  const float* enc_out_w = (const float*)d_in[5];
  const float* enc_out_b = (const float*)d_in[6];
  const float* enc_ff_w  = (const float*)d_in[7];
  const float* enc_ff_b  = (const float*)d_in[8];
  const float* pma_lin_w = (const float*)d_in[9];
  const float* pma_lin_b = (const float*)d_in[10];
  const float* pma_seed  = (const float*)d_in[11];
  const float* pma_in_w  = (const float*)d_in[12];
  const float* pma_in_b  = (const float*)d_in[13];
  const float* pma_out_w = (const float*)d_in[14];
  const float* pma_out_b = (const float*)d_in[15];
  const float* pma_ff_w  = (const float*)d_in[16];
  const float* pma_ff_b  = (const float*)d_in[17];
  const float* dec_in_w  = (const float*)d_in[18];
  const float* dec_in_b  = (const float*)d_in[19];
  const float* dec_out_w = (const float*)d_in[20];
  const float* dec_out_b = (const float*)d_in[21];
  const float* dec_ff_w  = (const float*)d_in[22];
  const float* dec_ff_b  = (const float*)d_in[23];
  const int*   edge_src  = (const int*)d_in[24];
  (void)in_sizes; (void)n_in; (void)out_size; (void)ws_size;

  float*    hfeat = (float*)d_ws;                      // NN*64 f32
  _Float16* wt    = (_Float16*)(hfeat + (size_t)NN * DH);  // 40960 halves
  float*    qseed = (float*)(wt + 40960);              // 64 f32
  float*    outp  = (float*)d_out;

  hipLaunchKernelGGL(prep_kernel, dim3(1), dim3(256), 0, stream,
                     w_lin, enc_in_w, enc_out_w, enc_ff_w, pma_lin_w, pma_in_w,
                     pma_in_b, pma_seed, wt, qseed);
  hipLaunchKernelGGL(lin1_kernel, dim3(NN / 256), dim3(256), 0, stream,
                     x, b_lin, wt, hfeat);
  hipLaunchKernelGGL(set_enc_kernel, dim3(NN), dim3(32), 0, stream,
                     hfeat, wt, qseed, edge_src,
                     enc_in_b, enc_out_b, enc_ff_b, pma_lin_b, pma_in_b, pma_seed,
                     pma_out_w, pma_out_b, pma_ff_w, pma_ff_b,
                     dec_in_w, dec_in_b, dec_out_w, dec_out_b, dec_ff_w, dec_ff_b,
                     outp);
}
